// RGATLayer_27075473834504
// MI455X (gfx1250) — matrix operand order repaired and checked by reading, not compiled, not run on hardware
//
#include <hip/hip_runtime.h>
#include <hip/hip_bf16.h>
#include <stdint.h>

typedef __attribute__((ext_vector_type(16))) _Float16 v16h;
typedef __attribute__((ext_vector_type(8)))  _Float16 v8h;
typedef __attribute__((ext_vector_type(8)))  float    v8f;

#define B_      8
#define N_      2048
#define M_      (B_ * N_)      // 16384 flattened rows
#define IN_DIM  512
#define OUT_DIM 256
#define ALPHA_  0.1f
#define NEG_INF_ (-9.0e15f)
#define LDS_STRIDE 40   // halves; 80B rows keep 16B alignment for b128 ds loads
#define LDS_TILE   (OUT_DIM * LDS_STRIDE)   // halves per buffer

// A-operand (16x32 f16) K index for half slot `idx` (0..15) in lane group g (0/1).
// Lanes 0-15 hold K=0..15, lanes 16-31 hold K=16..31 (contiguous).
__device__ __forceinline__ int kmapA(int idx, int g) {
    return idx + g * 16;
}

union BU { v16h v; v8h h[2]; };

__device__ __forceinline__ BU loadB16(const _Float16* bp) {
    BU u;
    u.h[0] = *(const v8h*)(bp);
    u.h[1] = *(const v8h*)(bp + 8);
    return u;
}

// CDNA5 async global->LDS copy, 16 bytes per lane. Tracked by ASYNCcnt.
__device__ __forceinline__ void async_copy_b128(uint32_t lds_byte_addr, const void* gptr) {
    asm volatile("global_load_async_to_lds_b128 %0, %1, off"
                 :: "v"(lds_byte_addr), "v"((uint64_t)(uintptr_t)gptr)
                 : "memory");
}
__device__ __forceinline__ void wait_asynccnt0() {
    asm volatile("s_wait_asynccnt 0x0" ::: "memory");
}

// ---------------------------------------------------------------------------
// prep: W (512x256 f32) -> WhT (256x512 f16, n-major), er[2] = rel_emb @ a3
// ---------------------------------------------------------------------------
__global__ __launch_bounds__(256) void prep_kernel(
    const float* __restrict__ W, const float* __restrict__ a,
    const float* __restrict__ rel, _Float16* __restrict__ WhT,
    float* __restrict__ er)
{
    int bid = blockIdx.x;
    if (bid < (IN_DIM * OUT_DIM) / 256) {
        int idx = bid * 256 + threadIdx.x;
        int k = idx / OUT_DIM, n = idx % OUT_DIM;
        WhT[(size_t)n * IN_DIM + k] = (_Float16)W[(size_t)k * OUT_DIM + n];
    } else if (threadIdx.x < 2) {
        float s = 0.f;
        for (int n = 0; n < OUT_DIM; ++n)
            s += rel[threadIdx.x * OUT_DIM + n] * a[2 * OUT_DIM + n];
        er[threadIdx.x] = s;
    }
}

// ---------------------------------------------------------------------------
// gemm1: h = x @ W  (M=16384, K=512, N=256), f16 WMMA, f32 accum.
// Each wave computes a 16x256 tile. Epilogue writes h TRANSPOSED
// (hT_glob[n][Mflat], n-major) with packed 16B stores, plus e_i / e_j.
// ---------------------------------------------------------------------------
__global__ __launch_bounds__(256) void gemm1_kernel(
    const float* __restrict__ x, const _Float16* __restrict__ WhT,
    const float* __restrict__ a, _Float16* __restrict__ hT_glob,
    float* __restrict__ ei, float* __restrict__ ej)
{
    int tid  = threadIdx.x;
    int wave = tid >> 5;
    int lane = tid & 31;
    int r = lane & 15, g = lane >> 4, g8 = g * 8;
    int rowbase = (blockIdx.x * 8 + wave) * 16;

    v8f c[16] = {};
    const float* xrow = x + (size_t)(rowbase + r) * IN_DIM;

    #pragma unroll 1
    for (int kt = 0; kt < IN_DIM / 32; ++kt) {
        if (kt + 1 < IN_DIM / 32)
            __builtin_prefetch(xrow + (kt + 1) * 32, 0, 0);

        v16h A;
        #pragma unroll
        for (int idx = 0; idx < 16; ++idx)
            A[idx] = (_Float16)xrow[kt * 32 + kmapA(idx, g)];

        const _Float16* bbase = WhT + (size_t)r * IN_DIM + kt * 32 + g * 16;
        BU Bc = loadB16(bbase);                       // nt = 0
        #pragma unroll
        for (int nt = 0; nt < 16; ++nt) {
            BU Bn;
            if (nt < 15) Bn = loadB16(bbase + (size_t)(nt + 1) * 16 * IN_DIM);
            c[nt] = __builtin_amdgcn_wmma_f32_16x16x32_f16(
                false, A, false, Bc.v, (short)0, c[nt], false, false);
            if (nt < 15) Bc = Bn;
        }
    }

    // store h transposed: element (row, n) -> hT_glob[n*M + row].
    // For fixed nt, the lane's 8 accumulator rows are contiguous -> one v8h store.
    #pragma unroll
    for (int nt = 0; nt < 16; ++nt) {
        v8h pk;
        #pragma unroll
        for (int i = 0; i < 8; ++i) pk[i] = (_Float16)c[nt][i];
        *(v8h*)(hT_glob + (size_t)(nt * 16 + r) * M_ + rowbase + g8) = pk;
    }

    // e_i = h . a1 , e_j = h . a2  (reduce across the 16 lanes of each half-wave)
    #pragma unroll
    for (int i = 0; i < 8; ++i) {
        float s1 = 0.f, s2 = 0.f;
        #pragma unroll
        for (int nt = 0; nt < 16; ++nt) {
            float hv = c[nt][i];
            int n = nt * 16 + r;
            s1 += hv * a[n];
            s2 += hv * a[OUT_DIM + n];
        }
        #pragma unroll
        for (int off = 1; off < 16; off <<= 1) {
            s1 += __shfl_xor(s1, off, 32);
            s2 += __shfl_xor(s2, off, 32);
        }
        if (r == 0) {
            ei[rowbase + i + g8] = s1;
            ej[rowbase + i + g8] = s2;
        }
    }
}

// ---------------------------------------------------------------------------
// attn: fused masked-softmax + attention@h (flash-attention style).
// 128 blocks = B * (N/128); 8 waves/block, wave owns 16 query rows.
// h tiles staged LDS-direct via global_load_async_to_lds_b128 (double-buffered);
// adj/s_mask prefetched one tile ahead (global_prefetch_b8).
// ---------------------------------------------------------------------------
__global__ __launch_bounds__(256) void attn_kernel(
    const int* __restrict__ adj, const int* __restrict__ sm32,
    const _Float16* __restrict__ hT_glob, const float* __restrict__ ei_g,
    const float* __restrict__ ej_g, const float* __restrict__ er_g,
    float* __restrict__ out)
{
    // two transposed h-tile buffers: hT[buf][n][k], stride 40 halves
    __shared__ _Float16 hT[2 * LDS_TILE];

    int tid  = threadIdx.x;
    int wave = tid >> 5, lane = tid & 31;
    int r = lane & 15, g = lane >> 4, g8 = g * 8;
    int b      = blockIdx.x >> 4;   // 16 row-blocks per batch
    int rowblk = blockIdx.x & 15;
    int qbase  = rowblk * 128 + wave * 16;
    int qr     = qbase + r;         // this lane's query row (A-matrix row)

    float er0 = er_g[0], er1 = er_g[1];
    float eival = ei_g[b * N_ + qr];
    const int*   adjrow = adj  + ((size_t)b * N_ + qr) * N_;
    const int*   smrow  = sm32 + (((size_t)b * N_ + qr) * N_) * 2; // i64 low dword
    const float* ejb    = ej_g + b * N_;

    // async-stage role: thread t owns LDS row n = t (4 x 16B chunks per tile)
    const _Float16* hrow = hT_glob + (size_t)tid * M_ + b * N_;
    uint32_t lds_row0 = (uint32_t)(uintptr_t)(&hT[0]) + (uint32_t)tid * (LDS_STRIDE * 2);

    float m = -__builtin_inff();
    float l = 0.f;
    v8f acc[16] = {};

    // preload tile 0 into buffer 0
    #pragma unroll
    for (int s2 = 0; s2 < 4; ++s2)
        async_copy_b128(lds_row0 + s2 * 16, hrow + s2 * 8);
    wait_asynccnt0();
    __syncthreads();

    int buf = 0;
    #pragma unroll 1
    for (int jb = 0; jb < N_; jb += 32) {
        bool have_next = (jb + 32 < N_);

        // prefetch next tile's adj / s_mask rows (HBM-bound gathers)
        if (have_next) {
            __builtin_prefetch(adjrow + (jb + 32) + g * 16, 0, 0);
            __builtin_prefetch(smrow + 2 * (jb + 32) + g * 32, 0, 0);
        }

        // masked leaky-relu scores for this lane's row, 16 K-slots
        float s[16];
        float tmax = NEG_INF_;
        #pragma unroll
        for (int idx = 0; idx < 16; ++idx) {
            int col = jb + kmapA(idx, g);
            int av  = adjrow[col];
            int sv  = smrow[2 * col];
            float e = eival + ejb[col] + (sv ? er1 : er0);
            e = e > 0.f ? e : ALPHA_ * e;
            float sc = (av > 0) ? e : NEG_INF_;
            s[idx] = sc;
            tmax = fmaxf(tmax, sc);
        }
        tmax = fmaxf(tmax, __shfl_xor(tmax, 16, 32));

        float mnew  = fmaxf(m, tmax);
        float scale = expf(m - mnew);

        // rescale running accumulators (row of element i is i+g8)
        #pragma unroll
        for (int i = 0; i < 8; ++i) {
            float sci = __shfl(scale, i + g8, 32);
            #pragma unroll
            for (int nt = 0; nt < 16; ++nt) acc[nt][i] *= sci;
        }

        // probabilities -> f16 A-operand, accumulate row sum
        v16h p;
        float lsum = 0.f;
        #pragma unroll
        for (int idx = 0; idx < 16; ++idx) {
            float pv = expf(s[idx] - mnew);
            lsum += pv;
            p[idx] = (_Float16)pv;
        }
        lsum += __shfl_xor(lsum, 16, 32);
        l = l * scale + lsum;
        m = mnew;

        // kick off async DMA of the next h tile into the other LDS buffer;
        // it overlaps with the 16-WMMA chain below.
        if (have_next) {
            uint32_t ldst = lds_row0 + (buf ^ 1) * (LDS_TILE * 2);
            const _Float16* srcr = hrow + (jb + 32);
            #pragma unroll
            for (int s2 = 0; s2 < 4; ++s2)
                async_copy_b128(ldst + s2 * 16, srcr + s2 * 8);
        }

        // acc += P (16x32) @ h_tile (32x256): 16 WMMAs, B from LDS
        {
            const _Float16* hTb = &hT[buf * LDS_TILE];
            const _Float16* bbase = hTb + (size_t)r * LDS_STRIDE + g * 16;
            BU Bc = loadB16(bbase);                   // nt = 0
            #pragma unroll
            for (int nt = 0; nt < 16; ++nt) {
                BU Bn;
                if (nt < 15) Bn = loadB16(bbase + (size_t)(nt + 1) * 16 * LDS_STRIDE);
                acc[nt] = __builtin_amdgcn_wmma_f32_16x16x32_f16(
                    false, p, false, Bc.v, (short)0, acc[nt], false, false);
                if (nt < 15) Bc = Bn;
            }
        }

        if (have_next) {
            wait_asynccnt0();       // our 4 chunks have landed in LDS
            __syncthreads();        // everyone's chunks have landed
            buf ^= 1;
        }
    }

    // normalize + ELU + store
    #pragma unroll
    for (int i = 0; i < 8; ++i) {
        float li   = __shfl(l, i + g8, 32);
        float rinv = 1.0f / li;
        int row = qbase + i + g8;
        float* orow = out + ((size_t)b * N_ + row) * OUT_DIM;
        #pragma unroll
        for (int nt = 0; nt < 16; ++nt) {
            float v = acc[nt][i] * rinv;
            v = v > 0.f ? v : expm1f(v);
            orow[nt * 16 + r] = v;
        }
    }
}

// ---------------------------------------------------------------------------
extern "C" void kernel_launch(void* const* d_in, const int* in_sizes, int n_in,
                              void* d_out, int out_size, void* d_ws, size_t ws_size,
                              hipStream_t stream) {
    (void)in_sizes; (void)n_in; (void)out_size; (void)ws_size;
    const float* x    = (const float*)d_in[0];
    const int*   adj  = (const int*)d_in[1];
    const int*   sm32 = (const int*)d_in[2];   // int64 mask, read low dwords
    const float* W    = (const float*)d_in[3];
    const float* a    = (const float*)d_in[4];
    const float* rel  = (const float*)d_in[5];
    float* out = (float*)d_out;

    char* ws = (char*)d_ws;
    _Float16* hT_glob = (_Float16*)ws; ws += (size_t)OUT_DIM * M_ * 2;     // 8 MB
    _Float16* WhT     = (_Float16*)ws; ws += (size_t)IN_DIM * OUT_DIM * 2; // 256 KB
    float* ei = (float*)ws;            ws += (size_t)M_ * 4;
    float* ej = (float*)ws;            ws += (size_t)M_ * 4;
    float* er = (float*)ws;            ws += 256;

    prep_kernel<<<(IN_DIM * OUT_DIM) / 256 + 1, 256, 0, stream>>>(W, a, rel, WhT, er);
    gemm1_kernel<<<M_ / 128, 256, 0, stream>>>(x, WhT, a, hT_glob, ei, ej);
    attn_kernel<<<B_ * (N_ / 128), 256, 0, stream>>>(adj, sm32, hT_glob, ei, ej, er, out);
}
